// GAT_7507602833709
// MI455X (gfx1250) — matrix operand order repaired and checked by reading, not compiled, not run on hardware
//
#include <hip/hip_runtime.h>
#include <hip/hip_bf16.h>
#include <math.h>

typedef __attribute__((ext_vector_type(16))) _Float16 v16h;
typedef __attribute__((ext_vector_type(8)))  _Float16 v8h;
typedef __attribute__((ext_vector_type(4)))  _Float16 v4h;
typedef __attribute__((ext_vector_type(8)))  float    v8f;
typedef __attribute__((ext_vector_type(4)))  float    v4f;
typedef __attribute__((ext_vector_type(2)))  float    v2f;
typedef __attribute__((ext_vector_type(2)))  int      v2i;

#define NN   4096   // nodes
#define FIN  256    // in features
#define NH   8      // heads
#define FO   64     // out features per head
#define HF   512    // NH*FO

static __device__ inline v8f wmma_f16(v16h a, v16h b, v8f c) {
    return __builtin_amdgcn_wmma_f32_16x16x32_f16(false, a, false, b, (short)0, c, false, false);
}

// async copy 8 bytes global -> LDS (per lane), tracked by ASYNCcnt
static __device__ inline void async_b64(const float* gsrc, float* ldst) {
    __builtin_amdgcn_global_load_async_to_lds_b64(
        (v2i*)gsrc,                                          // generic -> AS1 (device)
        (__attribute__((address_space(3))) v2i*)ldst,        // LDS
        0, 0);
}

// ---------------------------------------------------------------------------
// fp32 -> fp16 convert (vectorized x4)
// ---------------------------------------------------------------------------
__global__ void cvt_f16_kernel(const float* __restrict__ src,
                               _Float16* __restrict__ dst, int n4) {
    int i = blockIdx.x * 256 + threadIdx.x;
    if (i < n4) {
        v4f s = *(const v4f*)(src + i * 4);
        v4h d;
        d[0] = (_Float16)s[0]; d[1] = (_Float16)s[1];
        d[2] = (_Float16)s[2]; d[3] = (_Float16)s[3];
        *(v4h*)(dst + i * 4) = d;
    }
}

// ---------------------------------------------------------------------------
// projT = (x @ W^T)^T computed as W @ x^T : M=512 feats, N=4096 nodes, K=256.
// blockIdx.z==0 -> proj weights, store f16 projT[512][4096]
// blockIdx.z==1 -> skip weights, store f32 skipT[512][4096]
// block: 256 thr = 8 waves stacked in M (tile 128x64). grid (64, 4, 2).
// ---------------------------------------------------------------------------
__global__ __launch_bounds__(256)
void proj_gemm_kernel(const _Float16* __restrict__ Wh,
                      const _Float16* __restrict__ skipWh,
                      const _Float16* __restrict__ xh,
                      _Float16* __restrict__ projT,
                      float* __restrict__ skipT) {
    const int  wave  = threadIdx.x >> 5;
    const int  lane  = threadIdx.x & 31;
    const int  laneM = lane & 15;
    const bool hi    = lane >= 16;
    const bool is_skip = (blockIdx.z != 0);
    const _Float16* __restrict__ Wp = is_skip ? skipWh : Wh;

    const int nBase = blockIdx.x * 64;              // node cols
    const int mBase = blockIdx.y * 128 + wave * 16; // feature rows
    const int rowA  = mBase + laneM;

    v8f c[4] = {v8f{}, v8f{}, v8f{}, v8f{}};

    for (int k = 0; k < FIN; k += 32) {
        // A = W row (16x32 f16 layout): lanes<16 K {0..7,16..23}, lanes>=16 K {8..15,24..31}
        const int ka = k + (hi ? 8 : 0);
        v8h alo = *(const v8h*)(Wp + rowA * FIN + ka);
        v8h ahi = *(const v8h*)(Wp + rowA * FIN + ka + 16);
        v16h a;
        #pragma unroll
        for (int q = 0; q < 8; ++q) { a[q] = alo[q]; a[q + 8] = ahi[q]; }

        const int kb = k + (hi ? 8 : 0);
        #pragma unroll
        for (int t = 0; t < 4; ++t) {
            // B = x^T (32x16): lane holds node column, K {kb..kb+7, kb+16..kb+23}
            const int node = nBase + t * 16 + laneM;
            v8h blo = *(const v8h*)(xh + node * FIN + kb);
            v8h bhi = *(const v8h*)(xh + node * FIN + kb + 16);
            v16h b;
            #pragma unroll
            for (int q = 0; q < 8; ++q) { b[q] = blo[q]; b[q + 8] = bhi[q]; }
            c[t] = wmma_f16(a, b, c[t]);
        }
    }

    #pragma unroll
    for (int t = 0; t < 4; ++t) {
        const int col = nBase + t * 16 + laneM;
        #pragma unroll
        for (int v = 0; v < 8; ++v) {
            const int row = mBase + v + (hi ? 8 : 0);
            if (!is_skip) projT[row * NN + col] = (_Float16)c[t][v];
            else          skipT[row * NN + col] = c[t][v];
        }
    }
}

// ---------------------------------------------------------------------------
// s_src[h][n] = sum_f projT[h*64+f][n] * a_src[h][f]   (same for tgt)
// ---------------------------------------------------------------------------
__global__ __launch_bounds__(256)
void score_vec_kernel(const _Float16* __restrict__ projT,
                      const float* __restrict__ a_src,
                      const float* __restrict__ a_tgt,
                      float* __restrict__ s_src,
                      float* __restrict__ s_tgt) {
    const int h = blockIdx.x >> 4;
    const int n = (blockIdx.x & 15) * 256 + threadIdx.x;
    float as = 0.f, at = 0.f;
    #pragma unroll 8
    for (int f = 0; f < FO; ++f) {
        float v = (float)projT[(h * FO + f) * NN + n];
        as += v * a_src[h * FO + f];
        at += v * a_tgt[h * FO + f];
    }
    s_src[h * NN + n] = as;
    s_tgt[h * NN + n] = at;
}

// ---------------------------------------------------------------------------
// Fused flash-GAT attention. One block = 16 query nodes x all 8 heads.
// 8 waves, wave w == head w. Online softmax in registers, P.V via WMMA.
// Mask tiles (16x32 f32) double-buffered in LDS via async-to-LDS copies,
// shared by all heads; one barrier per j-tile.
// ---------------------------------------------------------------------------
#define MSTRIDE 36   // padded LDS row stride (floats) to dodge bank conflicts

__global__ __launch_bounds__(256)
void attn_kernel(const float* __restrict__ mask,     // [4096][4096]
                 const _Float16* __restrict__ projT, // [512][4096]
                 const float* __restrict__ skipT,    // [512][4096]
                 const float* __restrict__ s_src,    // [8][4096]
                 const float* __restrict__ s_tgt,    // [8][4096]
                 const float* __restrict__ bias,     // [512]
                 float* __restrict__ out) {          // [4096][512]
    __shared__ float lds_mask[2][16 * MSTRIDE];

    const int  tid    = threadIdx.x;
    const int  h      = tid >> 5;                    // wave == head
    const int  lane   = tid & 31;
    const int  laneM  = lane & 15;
    const bool hihalf = lane >= 16;
    const int  iBase  = blockIdx.x * 16;

    // this thread's 2-float chunk of the mask tile (same for every tile)
    const int mr = tid >> 4;           // 0..15
    const int mc = (tid & 15) * 2;     // 0..30
    const float* mrow = mask + (iBase + mr) * NN + mc;

    const float ssrc = s_src[h * NN + iBase + laneM];

    float m_row = -INFINITY;   // running row max (valid per lane pair)
    float l_row = 0.f;         // running row sum

    v8f c[4] = {v8f{}, v8f{}, v8f{}, v8f{}};

    const int jj0 = hihalf ? 8 : 0;
    const int NT = NN / 32;

    // prologue: kick off tile 0
    async_b64(mrow, &lds_mask[0][mr * MSTRIDE + mc]);

    for (int jt = 0; jt < NT; ++jt) {
        const int jBase = jt * 32;
        const float* lm = lds_mask[jt & 1];

        // tile jt landed (this wave's chunk) ...
        asm volatile("s_wait_asynccnt 0x0" ::: "memory");
        // ... and everyone's chunk landed; everyone done reading tile jt-1
        __syncthreads();
        // overlap: fetch tile jt+1 into the other buffer
        if (jt + 1 < NT)
            async_b64(mrow + (jBase + 32),
                      &lds_mask[(jt + 1) & 1][mr * MSTRIDE + mc]);

        // ---- scores for this lane's 16 (row, j) elements (A-matrix order) ----
        float sc[16];
        {
            v4f t0 = *(const v4f*)(s_tgt + h * NN + jBase + jj0);
            v4f t1 = *(const v4f*)(s_tgt + h * NN + jBase + jj0 + 4);
            v4f t2 = *(const v4f*)(s_tgt + h * NN + jBase + 16 + jj0);
            v4f t3 = *(const v4f*)(s_tgt + h * NN + jBase + 16 + jj0 + 4);
            v4f m0 = *(const v4f*)&lm[laneM * MSTRIDE + jj0];
            v4f m1 = *(const v4f*)&lm[laneM * MSTRIDE + jj0 + 4];
            v4f m2 = *(const v4f*)&lm[laneM * MSTRIDE + 16 + jj0];
            v4f m3 = *(const v4f*)&lm[laneM * MSTRIDE + 16 + jj0 + 4];
            #pragma unroll
            for (int q = 0; q < 4; ++q) {
                float e;
                e = ssrc + t0[q]; e = e > 0.f ? e : 0.2f * e; sc[q]      = e + m0[q];
                e = ssrc + t1[q]; e = e > 0.f ? e : 0.2f * e; sc[q + 4]  = e + m1[q];
                e = ssrc + t2[q]; e = e > 0.f ? e : 0.2f * e; sc[q + 8]  = e + m2[q];
                e = ssrc + t3[q]; e = e > 0.f ? e : 0.2f * e; sc[q + 12] = e + m3[q];
            }
        }

        // ---- online softmax (row split across lane pair L, L+16) ----
        float mx = sc[0];
        #pragma unroll
        for (int q = 1; q < 16; ++q) mx = fmaxf(mx, sc[q]);
        mx = fmaxf(mx, __shfl_xor(mx, 16, 32));
        const float newm  = fmaxf(m_row, mx);
        const float scale = __expf(m_row - newm);   // exp(-inf)=0 on first tile

        float psum = 0.f;
        float p[16];
        #pragma unroll
        for (int q = 0; q < 16; ++q) { p[q] = __expf(sc[q] - newm); psum += p[q]; }
        psum += __shfl_xor(psum, 16, 32);
        l_row = l_row * scale + psum;
        m_row = newm;

        // ---- rescale accumulators: row of accum vgpr v is v (+8 for hi half) ----
        #pragma unroll
        for (int v = 0; v < 8; ++v) {
            const float sv = __shfl(scale, (hihalf ? 8 : 0) + v, 32);
            #pragma unroll
            for (int t = 0; t < 4; ++t) c[t][v] *= sv;
        }

        // ---- P tile as WMMA A operand ----
        v16h a;
        #pragma unroll
        for (int q = 0; q < 16; ++q) a[q] = (_Float16)p[q];

        // ---- V tiles from projT (B operand: lane = feature column, j {jb..jb+7, jb+16..jb+23}) ----
        const int jb = jBase + (hihalf ? 8 : 0);
        #pragma unroll
        for (int t = 0; t < 4; ++t) {
            const int feat = h * FO + t * 16 + laneM;
            v8h blo = *(const v8h*)(projT + feat * NN + jb);
            v8h bhi = *(const v8h*)(projT + feat * NN + jb + 16);
            v16h b;
            #pragma unroll
            for (int q = 0; q < 8; ++q) { b[q] = blo[q]; b[q + 8] = bhi[q]; }
            c[t] = wmma_f16(a, b, c[t]);
        }
    }

    // ---- epilogue: 1/l, + skip, + bias, ELU, store ----
    const float linv = 1.0f / l_row;
    #pragma unroll
    for (int t = 0; t < 4; ++t) {
        const int feat = h * FO + t * 16 + laneM;
        const float bv = bias[feat];
        const int rbase = iBase + (hihalf ? 8 : 0);
        v4f s0 = *(const v4f*)(skipT + feat * NN + rbase);
        v4f s1 = *(const v4f*)(skipT + feat * NN + rbase + 4);
        #pragma unroll
        for (int v = 0; v < 8; ++v) {
            const float li = __shfl(linv, (hihalf ? 8 : 0) + v, 32);
            float o = c[t][v] * li + (v < 4 ? s0[v] : s1[v - 4]) + bv;
            o = o > 0.f ? o : (__expf(o) - 1.0f);
            out[(rbase + v) * HF + feat] = o;
        }
    }
}

// ---------------------------------------------------------------------------
extern "C" void kernel_launch(void* const* d_in, const int* in_sizes, int n_in,
                              void* d_out, int out_size, void* d_ws, size_t ws_size,
                              hipStream_t stream) {
    const float* x      = (const float*)d_in[0];   // [4096,256]
    const float* mask   = (const float*)d_in[1];   // [4096,4096]
    const float* W      = (const float*)d_in[2];   // [512,256]
    const float* a_src  = (const float*)d_in[3];   // [1,8,64]
    const float* a_tgt  = (const float*)d_in[4];   // [1,8,64]
    const float* skip_W = (const float*)d_in[5];   // [512,256]
    const float* bias   = (const float*)d_in[6];   // [512]
    float* out = (float*)d_out;

    char* ws = (char*)d_ws;
    size_t off = 0;
    _Float16* xh     = (_Float16*)(ws + off); off += (size_t)NN * FIN * 2;   // 2 MB
    _Float16* Wh     = (_Float16*)(ws + off); off += (size_t)HF * FIN * 2;   // 256 KB
    _Float16* skipWh = (_Float16*)(ws + off); off += (size_t)HF * FIN * 2;   // 256 KB
    _Float16* projT  = (_Float16*)(ws + off); off += (size_t)HF * NN * 2;    // 4 MB
    float*    skipT  = (float*)   (ws + off); off += (size_t)HF * NN * 4;    // 8 MB
    float*    s_src  = (float*)   (ws + off); off += (size_t)NH * NN * 4;
    float*    s_tgt  = (float*)   (ws + off); off += (size_t)NH * NN * 4;

    // 1) fp32 -> fp16
    cvt_f16_kernel<<<(NN * FIN / 4 + 255) / 256, 256, 0, stream>>>(x, xh, NN * FIN / 4);
    cvt_f16_kernel<<<(HF * FIN / 4 + 255) / 256, 256, 0, stream>>>(W, Wh, HF * FIN / 4);
    cvt_f16_kernel<<<(HF * FIN / 4 + 255) / 256, 256, 0, stream>>>(skip_W, skipWh, HF * FIN / 4);

    // 2) proj^T (f16) and skip^T (f32) via WMMA
    proj_gemm_kernel<<<dim3(NN / 64, HF / 128, 2), 256, 0, stream>>>(Wh, skipWh, xh, projT, skipT);

    // 3) per-node attention scalars
    score_vec_kernel<<<NH * (NN / 256), 256, 0, stream>>>(projT, a_src, a_tgt, s_src, s_tgt);

    // 4) fused flash-GAT (softmax + async-LDS mask pipeline + WMMA + ELU)
    attn_kernel<<<NN / 16, 256, 0, stream>>>(mask, projT, skipT, s_src, s_tgt, bias, out);
}
